// AttentionSW_25185688223969
// MI455X (gfx1250) — matrix operand order repaired and checked by reading, not compiled, not run on hardware
//
#include <hip/hip_runtime.h>
#include <stdint.h>

// ---------------- types for WMMA / TDM ----------------
typedef __attribute__((ext_vector_type(16))) __bf16 v16bf;
typedef __attribute__((ext_vector_type(8)))  float  v8f;
typedef __attribute__((ext_vector_type(4)))  unsigned int u32x4;
typedef __attribute__((ext_vector_type(8)))  int i32x8;
typedef __attribute__((ext_vector_type(4)))  int i32x4;

union Frag {
    v16bf v;
    uint4 q[2];
    unsigned short u[16];
};

// Problem constants
#define BB   256
#define NN   196
#define DIMD 384
#define HH   12
#define KDIM 32
#define FQKV 1152          // H*(2*KD+VD)
#define ROWS 50176         // B*N
#define MT   3136          // ROWS/16
#define NPAD 224           // padded key/query length, 7*32 = 14*16
#define SW   224

// =================== 1. LayerNorm -> bf16 ===================
__global__ __launch_bounds__(256)
void ln_kernel(const float* __restrict__ x, const float* __restrict__ w,
               const float* __restrict__ bias, __bf16* __restrict__ xn) {
    int wid = threadIdx.x >> 5, lane = threadIdx.x & 31;
    int row = blockIdx.x * 8 + wid;
    if (row >= ROWS) return;
    const float* xr = x + (long)row * DIMD;
    float v[12];
    float s = 0.f;
    #pragma unroll
    for (int t = 0; t < 12; ++t) { v[t] = xr[t * 32 + lane]; s += v[t]; }
    #pragma unroll
    for (int m = 16; m >= 1; m >>= 1) s += __shfl_xor(s, m, 32);
    float mu = s * (1.f / 384.f);
    float s2 = 0.f;
    #pragma unroll
    for (int t = 0; t < 12; ++t) { float d = v[t] - mu; s2 += d * d; }
    #pragma unroll
    for (int m = 16; m >= 1; m >>= 1) s2 += __shfl_xor(s2, m, 32);
    float r = rsqrtf(s2 * (1.f / 384.f) + 1e-5f);
    __bf16* o = xn + (long)row * DIMD;
    #pragma unroll
    for (int t = 0; t < 12; ++t) {
        int e = t * 32 + lane;
        o[e] = (__bf16)((v[t] - mu) * r * w[e] + bias[e]);
    }
}

// =================== 2. f32 -> bf16 weight convert ===================
__global__ __launch_bounds__(256)
void cvt_kernel(const float* __restrict__ in, __bf16* __restrict__ out, int n) {
    int i = blockIdx.x * 256 + threadIdx.x;
    if (i < n) out[i] = (__bf16)in[i];
}

// =================== 3. bias gather expand: bias_full[h][n][m] ===================
__global__ __launch_bounds__(256)
void bias_kernel(const float* __restrict__ ab, const int* __restrict__ idxs,
                 float* __restrict__ bias_full) {
    int i = blockIdx.x * 256 + threadIdx.x;
    if (i >= HH * NN * NN) return;
    int nm = i % (NN * NN);
    int h  = i / (NN * NN);
    bias_full[i] = ab[h * NN + idxs[nm]];
}

// =================== 4. QKV GEMM: wave computes 16x64 tile ===================
__global__ __launch_bounds__(256)
void qkv_gemm_kernel(const __bf16* __restrict__ xn, const __bf16* __restrict__ wq,
                     const float* __restrict__ qkv_b,
                     __bf16* __restrict__ qb, __bf16* __restrict__ kb,
                     __bf16* __restrict__ vb) {
    const int lane = threadIdx.x & 31, wid = threadIdx.x >> 5;
    long tile = (long)blockIdx.x * 8 + wid;
    const int FT = FQKV / 64;                 // 18 column groups of 64
    int tf = (int)(tile % FT);
    int tm = (int)(tile / FT);
    if (tm >= MT) return;
    int arow = tm * 16 + (lane & 15);
    const uint4* ap = (const uint4*)(xn + (long)arow * DIMD);
    const uint4* bp0 = (const uint4*)(wq + (long)(tf * 64 + 0 * 16 + (lane & 15)) * DIMD);
    const uint4* bp1 = (const uint4*)(wq + (long)(tf * 64 + 1 * 16 + (lane & 15)) * DIMD);
    const uint4* bp2 = (const uint4*)(wq + (long)(tf * 64 + 2 * 16 + (lane & 15)) * DIMD);
    const uint4* bp3 = (const uint4*)(wq + (long)(tf * 64 + 3 * 16 + (lane & 15)) * DIMD);
    const int ha = (lane >> 4) * 16;          // A: per-half K sub-offset
    const int hb = (lane >> 4) * 16;          // B: lanes16-31 take K+16
    v8f acc0 = {}, acc1 = {}, acc2 = {}, acc3 = {};
    #pragma unroll
    for (int kk = 0; kk < DIMD; kk += 32) {
        Frag a;
        a.q[0] = ap[(kk + ha) >> 3];
        a.q[1] = ap[(kk + ha + 8) >> 3];
        int bi = (kk + hb) >> 3;
        Frag b0, b1, b2, b3;
        b0.q[0] = bp0[bi]; b0.q[1] = bp0[bi + 1];
        b1.q[0] = bp1[bi]; b1.q[1] = bp1[bi + 1];
        b2.q[0] = bp2[bi]; b2.q[1] = bp2[bi + 1];
        b3.q[0] = bp3[bi]; b3.q[1] = bp3[bi + 1];
        acc0 = __builtin_amdgcn_wmma_f32_16x16x32_bf16(false, a.v, false, b0.v, (short)0, acc0, false, false);
        acc1 = __builtin_amdgcn_wmma_f32_16x16x32_bf16(false, a.v, false, b1.v, (short)0, acc1, false, false);
        acc2 = __builtin_amdgcn_wmma_f32_16x16x32_bf16(false, a.v, false, b2.v, (short)0, acc2, false, false);
        acc3 = __builtin_amdgcn_wmma_f32_16x16x32_bf16(false, a.v, false, b3.v, (short)0, acc3, false, false);
    }
    // epilogue: scatter to q (scaled), k, v in [b][h][n][32] bf16
    const float scale = 0.17677669529663689f;    // 32^-0.5
    v8f accs[4] = {acc0, acc1, acc2, acc3};
    #pragma unroll
    for (int t = 0; t < 4; ++t) {
        int f = tf * 64 + t * 16 + (lane & 15);
        int h = f / 96, c = f % 96;
        float bia = qkv_b[f];
        #pragma unroll
        for (int r = 0; r < 8; ++r) {
            int row  = tm * 16 + r + ((lane >> 4) << 3);
            int bidx = row / NN, nidx = row % NN;
            float val = accs[t][r] + bia;
            long off = ((long)(bidx * HH + h) * NN + nidx) * KDIM;
            if (c < 32)       qb[off + c]        = (__bf16)(val * scale);
            else if (c < 64)  kb[off + (c - 32)] = (__bf16)val;
            else              vb[off + (c - 64)] = (__bf16)val;
        }
    }
}

// =================== 5. Attention per (b,h) ===================
#if __has_builtin(__builtin_amdgcn_tensor_load_to_lds)
#define USE_TDM 1
#else
#define USE_TDM 0
#endif

#if USE_TDM
__device__ __forceinline__ void tdm_load_tile(unsigned lds_off, const __bf16* gsrc) {
    // D# group0: count=1 | lds_addr | global_addr[56:0] | type=2
    unsigned long long ga = (unsigned long long)(const void*)gsrc;
    u32x4 g0 = { 1u, lds_off, (unsigned)ga,
                 (unsigned)((ga >> 32) & 0x01FFFFFFu) | (2u << 30) };
    // D# group1: data_size=2B; tensor 32 x 196; tile 32 x 224 (OOB rows -> 0);
    // tensor_dim0_stride = 32 elements
    i32x8 g1 = { (int)(1u << 16),        // workgroup_mask=0, data_size=1 (2B)
                 (int)(32u << 16),       // atomic_barrier_addr=0 | tensor_dim0=32 (lo16)
                 (int)(196u << 16),      // tensor_dim0 hi=0 | tensor_dim1=196 (lo16)
                 (int)(32u << 16),       // tensor_dim1 hi=0 | tile_dim0=32
                 (int)NPAD,              // tile_dim1=224 | tile_dim2=0
                 32,                     // tensor_dim0_stride lo32
                 0, 0 };                 // stride hi | tensor_dim1_stride (2D: unused)
    i32x4 z4 = { 0, 0, 0, 0 };
#if defined(__clang_major__) && (__clang_major__ >= 23)
    i32x8 z8 = { 0, 0, 0, 0, 0, 0, 0, 0 };
    __builtin_amdgcn_tensor_load_to_lds(g0, g1, z4, z4, z8, 0);
#else
    __builtin_amdgcn_tensor_load_to_lds(g0, g1, z4, z4, 0);
#endif
}
#endif

__global__ __launch_bounds__(256)
void attn_kernel(const __bf16* __restrict__ qb, const __bf16* __restrict__ kb,
                 const __bf16* __restrict__ vb, const float* __restrict__ bias_full,
                 __bf16* __restrict__ ao) {
    extern __shared__ char smem[];
    __bf16* qs = (__bf16*)smem;            // 224 x 32
    __bf16* ks = qs + NPAD * KDIM;
    __bf16* vs = ks + NPAD * KDIM;
    char* wbase = (char*)(vs + NPAD * KDIM);
    const int lane = threadIdx.x & 31, wid = threadIdx.x >> 5;
    const int WSTRIDE = 16 * SW * 4 + 16 * SW * 2 + 64;
    float*  S   = (float*)(wbase + wid * WSTRIDE);   // 16 x 224 f32 scores
    __bf16* P   = (__bf16*)(S + 16 * SW);            // 16 x 224 bf16 probs
    float*  INV = (float*)(P + 16 * SW);             // 16 x f32 1/rowsum

    int bh = blockIdx.x;
    int h = bh % HH, b = bh / HH;
    long base = (long)bh * NN * KDIM;

#if USE_TDM
    if (wid == 0) {   // one wave drives the Tensor Data Mover for all 3 tiles
        tdm_load_tile((unsigned)(size_t)(void*)qs, qb + base);
        tdm_load_tile((unsigned)(size_t)(void*)ks, kb + base);
        tdm_load_tile((unsigned)(size_t)(void*)vs, vb + base);
        __builtin_amdgcn_s_wait_tensorcnt(0);
    }
#else
    for (int idx = threadIdx.x; idx < 3 * NPAD; idx += 256) {
        int mat = idx / NPAD, row = idx % NPAD;
        __bf16* dstb = (mat == 0 ? qs : (mat == 1 ? ks : vs)) + row * KDIM;
        uint4* dst = (uint4*)dstb;
        if (row < NN) {
            const __bf16* srcb = (mat == 0 ? qb : (mat == 1 ? kb : vb)) + base + (long)row * KDIM;
            const uint4* src = (const uint4*)srcb;
            dst[0] = src[0]; dst[1] = src[1]; dst[2] = src[2]; dst[3] = src[3];
        } else {
            uint4 z = make_uint4(0u, 0u, 0u, 0u);
            dst[0] = z; dst[1] = z; dst[2] = z; dst[3] = z;
        }
    }
#endif
    __syncthreads();

    const float* bh_bias = bias_full + (long)h * NN * NN;
    const int ha = (lane >> 4) * 16;
    const int hb = (lane >> 4) * 16;

    for (int i = wid; i < 13; i += 8) {          // 13 row tiles of 16 queries
        // ---- scores: S[16 x 208] = q_tile (16x32) x k^T, + bias ----
        int am = i * 16 + (lane & 15);
        Frag a;
        a.q[0] = *(const uint4*)(qs + am * KDIM + ha);
        a.q[1] = *(const uint4*)(qs + am * KDIM + ha + 8);
        for (int j = 0; j < 13; ++j) {
            int bn = j * 16 + (lane & 15);
            Frag bm;
            bm.q[0] = *(const uint4*)(ks + bn * KDIM + hb);
            bm.q[1] = *(const uint4*)(ks + bn * KDIM + hb + 8);
            v8f acc = {};
            acc = __builtin_amdgcn_wmma_f32_16x16x32_bf16(false, a.v, false, bm.v,
                                                          (short)0, acc, false, false);
            int ncol = j * 16 + (lane & 15);
            #pragma unroll
            for (int r = 0; r < 8; ++r) {
                int mrow = r + ((lane >> 4) << 3);
                int qrow = i * 16 + mrow;
                float sv = -1e30f;
                if (ncol < NN)
                    sv = acc[r] + ((qrow < NN) ? bh_bias[qrow * NN + ncol] : 0.f);
                S[mrow * SW + ncol] = sv;
            }
        }
        // ---- softmax: 2 lanes per row ----
        int row = lane & 15, half = lane >> 4;
        float mx = -1e30f;
        for (int c = 0; c < 104; ++c)
            mx = fmaxf(mx, S[row * SW + half * 104 + c]);
        mx = fmaxf(mx, __shfl_xor(mx, 16, 32));
        float sum = 0.f;
        for (int c = 0; c < 112; ++c) {
            int col = half * 112 + c;
            float p = 0.f;
            if (col < 208) { p = __expf(S[row * SW + col] - mx); sum += p; }
            P[row * SW + col] = (__bf16)p;
        }
        sum += __shfl_xor(sum, 16, 32);
        if (half == 0) INV[row] = 1.f / sum;
        // ---- PV: out(16x32) = P(16x224) x V(224x32), K in 7 steps of 32 ----
        v8f acc0 = {}, acc1 = {};
        int pm = lane & 15;
        int n0 = lane & 15;
        for (int kt = 0; kt < 7; ++kt) {
            int kbase = kt * 32;
            Frag pa;
            pa.q[0] = *(const uint4*)(P + pm * SW + kbase + ha);
            pa.q[1] = *(const uint4*)(P + pm * SW + kbase + ha + 8);
            Frag b0, b1;
            int kb16 = kbase + hb;
            #pragma unroll
            for (int e = 0; e < 16; ++e) {
                b0.u[e] = *(const unsigned short*)(vs + (kb16 + e) * KDIM + n0);
                b1.u[e] = *(const unsigned short*)(vs + (kb16 + e) * KDIM + n0 + 16);
            }
            acc0 = __builtin_amdgcn_wmma_f32_16x16x32_bf16(false, pa.v, false, b0.v,
                                                           (short)0, acc0, false, false);
            acc1 = __builtin_amdgcn_wmma_f32_16x16x32_bf16(false, pa.v, false, b1.v,
                                                           (short)0, acc1, false, false);
        }
        // ---- store (fold 1/rowsum), layout [b][n][h*32+vd] bf16 ----
        #pragma unroll
        for (int r = 0; r < 8; ++r) {
            int mrow = r + ((lane >> 4) << 3);
            int qrow = i * 16 + mrow;
            if (qrow < NN) {
                float iv = INV[mrow];
                long o = ((long)(b * NN + qrow)) * DIMD + h * KDIM;
                ao[o + (lane & 15)]      = (__bf16)(acc0[r] * iv);
                ao[o + 16 + (lane & 15)] = (__bf16)(acc1[r] * iv);
            }
        }
    }
}

// =================== 6. Proj GEMM: wave computes 16x64 tile, f32 out ===================
__global__ __launch_bounds__(256)
void proj_gemm_kernel(const __bf16* __restrict__ ao, const __bf16* __restrict__ pw,
                      const float* __restrict__ pb, float* __restrict__ out) {
    const int lane = threadIdx.x & 31, wid = threadIdx.x >> 5;
    long tile = (long)blockIdx.x * 8 + wid;
    const int DT = DIMD / 64;                 // 6 column groups of 64
    int td = (int)(tile % DT);
    int tm = (int)(tile / DT);
    if (tm >= MT) return;
    int arow = tm * 16 + (lane & 15);
    const uint4* ap = (const uint4*)(ao + (long)arow * DIMD);
    const uint4* bp0 = (const uint4*)(pw + (long)(td * 64 + 0 * 16 + (lane & 15)) * DIMD);
    const uint4* bp1 = (const uint4*)(pw + (long)(td * 64 + 1 * 16 + (lane & 15)) * DIMD);
    const uint4* bp2 = (const uint4*)(pw + (long)(td * 64 + 2 * 16 + (lane & 15)) * DIMD);
    const uint4* bp3 = (const uint4*)(pw + (long)(td * 64 + 3 * 16 + (lane & 15)) * DIMD);
    const int ha = (lane >> 4) * 16;
    const int hb = (lane >> 4) * 16;
    v8f acc0 = {}, acc1 = {}, acc2 = {}, acc3 = {};
    #pragma unroll
    for (int kk = 0; kk < DIMD; kk += 32) {
        Frag a;
        a.q[0] = ap[(kk + ha) >> 3];
        a.q[1] = ap[(kk + ha + 8) >> 3];
        int bi = (kk + hb) >> 3;
        Frag b0, b1, b2, b3;
        b0.q[0] = bp0[bi]; b0.q[1] = bp0[bi + 1];
        b1.q[0] = bp1[bi]; b1.q[1] = bp1[bi + 1];
        b2.q[0] = bp2[bi]; b2.q[1] = bp2[bi + 1];
        b3.q[0] = bp3[bi]; b3.q[1] = bp3[bi + 1];
        acc0 = __builtin_amdgcn_wmma_f32_16x16x32_bf16(false, a.v, false, b0.v, (short)0, acc0, false, false);
        acc1 = __builtin_amdgcn_wmma_f32_16x16x32_bf16(false, a.v, false, b1.v, (short)0, acc1, false, false);
        acc2 = __builtin_amdgcn_wmma_f32_16x16x32_bf16(false, a.v, false, b2.v, (short)0, acc2, false, false);
        acc3 = __builtin_amdgcn_wmma_f32_16x16x32_bf16(false, a.v, false, b3.v, (short)0, acc3, false, false);
    }
    v8f accs[4] = {acc0, acc1, acc2, acc3};
    #pragma unroll
    for (int t = 0; t < 4; ++t) {
        int d = td * 64 + t * 16 + (lane & 15);
        float bia = pb[d];
        #pragma unroll
        for (int r = 0; r < 8; ++r) {
            int row = tm * 16 + r + ((lane >> 4) << 3);
            out[(long)row * DIMD + d] = accs[t][r] + bia;
        }
    }
}

// =================== launch ===================
extern "C" void kernel_launch(void* const* d_in, const int* in_sizes, int n_in,
                              void* d_out, int out_size, void* d_ws, size_t ws_size,
                              hipStream_t stream) {
    const float* x      = (const float*)d_in[0];
    const float* norm_w = (const float*)d_in[1];
    const float* norm_b = (const float*)d_in[2];
    const float* qkv_w  = (const float*)d_in[3];
    const float* qkv_b  = (const float*)d_in[4];
    const float* ab     = (const float*)d_in[5];
    const float* proj_w = (const float*)d_in[6];
    const float* proj_b = (const float*)d_in[7];
    const int*   idxs   = (const int*)d_in[8];
    float* out = (float*)d_out;

    size_t off = 0;
    auto alloc = [&](size_t bytes) {
        void* p = (char*)d_ws + off;
        off += (bytes + 255) & ~(size_t)255;
        return p;
    };
    __bf16* xn        = (__bf16*)alloc((size_t)ROWS * DIMD * 2);
    __bf16* wq        = (__bf16*)alloc((size_t)FQKV * DIMD * 2);
    __bf16* pw        = (__bf16*)alloc((size_t)DIMD * DIMD * 2);
    float*  bias_full = (float*) alloc((size_t)HH * NN * NN * 4);
    __bf16* qb        = (__bf16*)alloc((size_t)BB * HH * NN * KDIM * 2);
    __bf16* kb        = (__bf16*)alloc((size_t)BB * HH * NN * KDIM * 2);
    __bf16* vb        = (__bf16*)alloc((size_t)BB * HH * NN * KDIM * 2);
    __bf16* ao        = (__bf16*)alloc((size_t)ROWS * DIMD * 2);

    ln_kernel<<<ROWS / 8, 256, 0, stream>>>(x, norm_w, norm_b, xn);

    int nwq = FQKV * DIMD, npw = DIMD * DIMD, nbf = HH * NN * NN;
    cvt_kernel<<<(nwq + 255) / 256, 256, 0, stream>>>(qkv_w, wq, nwq);
    cvt_kernel<<<(npw + 255) / 256, 256, 0, stream>>>(proj_w, pw, npw);
    bias_kernel<<<(nbf + 255) / 256, 256, 0, stream>>>(ab, idxs, bias_full);

    qkv_gemm_kernel<<<(MT * (FQKV / 64)) / 8, 256, 0, stream>>>(xn, wq, qkv_b, qb, kb, vb);

    const int WSTRIDE = 16 * SW * 4 + 16 * SW * 2 + 64;
    size_t smem = (size_t)3 * NPAD * KDIM * 2 + (size_t)8 * WSTRIDE;
    attn_kernel<<<BB * HH, 256, smem, stream>>>(qb, kb, vb, bias_full, ao);

    proj_gemm_kernel<<<(MT * (DIMD / 64)) / 8, 256, 0, stream>>>(ao, pw, proj_b, out);
}
